// GRUObservationCell_45097156608133
// MI455X (gfx1250) — matrix operand order repaired and checked by reading, not compiled, not run on hardware
//
#include <hip/hip_runtime.h>
#include <hip/hip_bf16.h>

// ---------------- problem constants ----------------
#define N_ROWS 100000
#define HID    512
#define DIM    64
#define PHID   16
#define NOBS   50000
#define TH3    1536          // 3*HID
#define KIH    1024          // DIM*PHID
#define KHH    512
#define RT     (NOBS/16)     // 3125 row tiles (exact)
#define KS_IH  (KIH/32)      // 32 K-steps
#define KS_HH  (KHH/32)      // 16 K-steps

#define A_IH_HALFS (KS_IH*32*16)   // 16384 halfs = 32 KB per row tile
#define A_HH_HALFS (KS_HH*32*16)   //  8192 halfs = 16 KB per row tile

typedef __attribute__((ext_vector_type(16))) __bf16 v16bf;
typedef __attribute__((ext_vector_type(8)))  float  v8f;
typedef __attribute__((ext_vector_type(4)))  float  v4f;
typedef __attribute__((ext_vector_type(4)))  int    v4i;

#if defined(__has_builtin)
#if __has_builtin(__builtin_amdgcn_global_load_async_to_lds_b128)
#define HAVE_ASYNC_LDS 1
#endif
#if __has_builtin(__builtin_amdgcn_s_wait_asynccnt)
#define HAVE_WAIT_ASYNC 1
#endif
#endif

// round-to-nearest-even f32 -> bf16 bits
__device__ __forceinline__ unsigned f2bf(float f) {
  unsigned u = __builtin_bit_cast(unsigned, f);
  return (u + 0x7fffu + ((u >> 16) & 1u)) >> 16;
}

// ---------------- tiny init ----------------
__global__ void init_loss_kernel(float* loss) { *loss = 0.0f; }

// ---------------- bulk copy h -> h_new (NT stores: pure stream-out) ----------------
__global__ __launch_bounds__(256) void copy_h_kernel(
    const v4f* __restrict__ src, v4f* __restrict__ dst, int n4) {
  int i = blockIdx.x * 256 + threadIdx.x;
  if (i < n4) {
    v4f v = __builtin_nontemporal_load(&src[i]);
    __builtin_nontemporal_store(v, &dst[i]);
  }
}

// ---------------- pack weights into WMMA B-fragment layout ----------------
// B (32x16, KxN) lane layout: lane l holds col n=l&15; lanes 0-15: K in {0..7,16..23},
// lanes 16-31: K in {8..15,24..31}. packed index: (((jt*KS + ks)*32 + lane)*16 + h)
__global__ __launch_bounds__(256) void pack_w_kernel(
    const float* __restrict__ W, unsigned short* __restrict__ dst,
    int K, int KS) {
  int k8n = K >> 3;
  int t = blockIdx.x * 256 + threadIdx.x;          // t = j*k8n + k8
  if (t >= TH3 * k8n) return;
  int j  = t / k8n;
  int k8 = t - j * k8n;
  int k0 = k8 * 8;
  int ks  = k0 >> 5;
  int off = k0 & 31;
  int lane = (j & 15) + ((off & 8) ? 16 : 0);
  int h0   = (off & 16) ? 8 : 0;                    // 0 or 8
  int jt   = j >> 4;
  const float* src = W + (size_t)j * K + k0;
  size_t base = (((size_t)jt * KS + ks) * 32 + lane) * 16 + h0;
  uint4 pk;
  pk.x = f2bf(src[0]) | (f2bf(src[1]) << 16);
  pk.y = f2bf(src[2]) | (f2bf(src[3]) << 16);
  pk.z = f2bf(src[4]) | (f2bf(src[5]) << 16);
  pk.w = f2bf(src[6]) | (f2bf(src[7]) << 16);
  *(uint4*)(dst + base) = pk;
}

// ---------------- prep: loss + gru_in written in packed A layout ----------------
// A (16x32, MxK) lane layout: lane l row = l&15; lanes 0-15 hold K in
// {0..7, 16..23}, lanes 16-31 hold K in {8..15, 24..31} (h<8 -> first range).
__global__ __launch_bounds__(256) void prep_kernel(
    const float* __restrict__ p, const float* __restrict__ X,
    const float* __restrict__ M, const int* __restrict__ iobs,
    const float* __restrict__ wprep, const float* __restrict__ bprep,
    unsigned short* __restrict__ packA_ih, float* __restrict__ loss) {
  int t = blockIdx.x * 256 + threadIdx.x;          // t = n*64 + d
  if (t >= NOBS * DIM) return;
  int n = t >> 6, d = t & 63;
  int row = iobs[n];
  float mean = p[(size_t)row * (2 * DIM) + d];
  float var  = fabsf(p[(size_t)row * (2 * DIM) + DIM + d]) + 1e-6f;
  float x = X[t];
  float m = M[t];
  float err = (x - mean) * rsqrtf(var);
  float lc = 0.5f * (err * err + __logf(var)) * m;
  #pragma unroll
  for (int o = 16; o > 0; o >>= 1) lc += __shfl_down(lc, o, 32);
  if ((threadIdx.x & 31) == 0) atomicAdd(loss, lc);

  const float* w = wprep + (size_t)d * 4 * PHID;   // [4][PH]
  float vals[PHID];
  #pragma unroll
  for (int j = 0; j < PHID; ++j) {
    float a = bprep[d * PHID + j]
            + x    * w[0 * PHID + j]
            + mean * w[1 * PHID + j]
            + var  * w[2 * PHID + j]
            + err  * w[3 * PHID + j];
    vals[j] = fmaxf(a, 0.0f) * m;
  }
  // scatter 16 contiguous K values (K base = d*16) into A-fragment layout
  int rt = n >> 4;
  int ks = d >> 1;
  int h0 = (d & 1) ? 8 : 0;     // d even -> K offsets 0..15; odd -> 16..31
  int lane_lo = n & 15;
  #pragma unroll
  for (int c = 0; c < 2; ++c) { // chunk0 -> lanes 0-15, chunk1 -> lanes 16-31
    int lane = lane_lo + c * 16;
    size_t base = (((size_t)rt * KS_IH + ks) * 32 + lane) * 16 + h0;
    uint4 pk;
    pk.x = f2bf(vals[c*8+0]) | (f2bf(vals[c*8+1]) << 16);
    pk.y = f2bf(vals[c*8+2]) | (f2bf(vals[c*8+3]) << 16);
    pk.z = f2bf(vals[c*8+4]) | (f2bf(vals[c*8+5]) << 16);
    pk.w = f2bf(vals[c*8+6]) | (f2bf(vals[c*8+7]) << 16);
    *(uint4*)(packA_ih + base) = pk;
  }
}

// ---------------- gather h[i_obs] into packed A layout (bf16) ----------------
__global__ __launch_bounds__(256) void gather_h_kernel(
    const float* __restrict__ h, const int* __restrict__ iobs,
    unsigned short* __restrict__ packA_hh) {
  int t = blockIdx.x * 256 + threadIdx.x;          // t = n*64 + c8
  if (t >= NOBS * (KHH / 8)) return;
  int n = t >> 6, c8 = t & 63;
  int row = iobs[n];
  const float4* src = (const float4*)(h + (size_t)row * HID + c8 * 8);
  float4 f0 = src[0], f1 = src[1];
  int rt = n >> 4;
  int k0 = c8 * 8;
  int ks  = k0 >> 5;
  int off = k0 & 31;
  int h0   = (off & 16) ? 8 : 0;
  int lane = (n & 15) + ((off & 8) ? 16 : 0);
  size_t base = (((size_t)rt * KS_HH + ks) * 32 + lane) * 16 + h0;
  uint4 pk;
  pk.x = f2bf(f0.x) | (f2bf(f0.y) << 16);
  pk.y = f2bf(f0.z) | (f2bf(f0.w) << 16);
  pk.z = f2bf(f1.x) | (f2bf(f1.y) << 16);
  pk.w = f2bf(f1.z) | (f2bf(f1.w) << 16);
  *(uint4*)(packA_hh + base) = pk;
}

// ---------------- fused gate GEMMs + GRU epilogue + scatter ----------------
// grid: (RT, 4); block: 256 = 8 waves; wave computes 16 rows x 16 cols.
// Packed A row-tile (32KB ih + 16KB hh) is staged in LDS once per workgroup
// via CDNA5 async global->LDS copies (all 8 waves share identical A frags).
// accR/accZ accumulate ih-part AND hh-part; accIN/accHN kept separate for
// n = tanh(inn + r*hn).
__global__ __launch_bounds__(256) void gates_kernel(
    const unsigned short* __restrict__ pAih,
    const unsigned short* __restrict__ pAhh,
    const unsigned short* __restrict__ pBih,
    const unsigned short* __restrict__ pBhh,
    const float* __restrict__ bih, const float* __restrict__ bhh,
    const float* __restrict__ h,  const int* __restrict__ iobs,
    float* __restrict__ out) {
  __shared__ unsigned short ldsA[A_IH_HALFS + A_HH_HALFS];  // 48 KB

  int rt   = blockIdx.x;
  int tid  = threadIdx.x;
  int wv   = tid >> 5;
  int lane = tid & 31;
  int col0 = blockIdx.y * 128 + wv * 16;
  int ct   = col0 >> 4;                            // 0..31 column tile in gate

  const unsigned short* gAih = pAih + (size_t)rt * A_IH_HALFS;
  const unsigned short* gAhh = pAhh + (size_t)rt * A_HH_HALFS;

#if defined(HAVE_ASYNC_LDS)
  {
    typedef __attribute__((address_space(1))) v4i* gp_t;
    typedef __attribute__((address_space(3))) v4i* lp_t;
    #pragma unroll
    for (int i = 0; i < A_IH_HALFS / (256 * 8); ++i) {     // 8 iters
      int c = (i * 256 + tid) * 8;
      __builtin_amdgcn_global_load_async_to_lds_b128(
          (gp_t)(gAih + c), (lp_t)(ldsA + c), 0, 0);
    }
    #pragma unroll
    for (int i = 0; i < A_HH_HALFS / (256 * 8); ++i) {     // 4 iters
      int c = (i * 256 + tid) * 8;
      __builtin_amdgcn_global_load_async_to_lds_b128(
          (gp_t)(gAhh + c), (lp_t)(ldsA + A_IH_HALFS + c), 0, 0);
    }
#if defined(HAVE_WAIT_ASYNC)
    __builtin_amdgcn_s_wait_asynccnt(0);
#else
    asm volatile("s_wait_asynccnt 0" ::: "memory");
#endif
  }
#else
  {
    #pragma unroll
    for (int i = 0; i < A_IH_HALFS / (256 * 8); ++i) {
      int c = (i * 256 + tid) * 8;
      *(uint4*)(ldsA + c) = *(const uint4*)(gAih + c);
    }
    #pragma unroll
    for (int i = 0; i < A_HH_HALFS / (256 * 8); ++i) {
      int c = (i * 256 + tid) * 8;
      *(uint4*)(ldsA + A_IH_HALFS + c) = *(const uint4*)(gAhh + c);
    }
  }
#endif
  __syncthreads();

  v8f accR = {}, accZ = {}, accIN = {}, accHN = {};

  {
    const v16bf* A  = (const v16bf*)ldsA + lane;
    const v16bf* BR = (const v16bf*)pBih + ((size_t)(ct)      * KS_IH) * 32 + lane;
    const v16bf* BZ = (const v16bf*)pBih + ((size_t)(32 + ct) * KS_IH) * 32 + lane;
    const v16bf* BN = (const v16bf*)pBih + ((size_t)(64 + ct) * KS_IH) * 32 + lane;
    #pragma unroll 4
    for (int ks = 0; ks < KS_IH; ++ks) {
      v16bf a  = A [ks * 32];
      v16bf br = BR[ks * 32];
      v16bf bz = BZ[ks * 32];
      v16bf bn = BN[ks * 32];
      accR  = __builtin_amdgcn_wmma_f32_16x16x32_bf16(false, a, false, br, (short)0, accR,  false, false);
      accZ  = __builtin_amdgcn_wmma_f32_16x16x32_bf16(false, a, false, bz, (short)0, accZ,  false, false);
      accIN = __builtin_amdgcn_wmma_f32_16x16x32_bf16(false, a, false, bn, (short)0, accIN, false, false);
    }
  }
  {
    const v16bf* A  = (const v16bf*)(ldsA + A_IH_HALFS) + lane;
    const v16bf* BR = (const v16bf*)pBhh + ((size_t)(ct)      * KS_HH) * 32 + lane;
    const v16bf* BZ = (const v16bf*)pBhh + ((size_t)(32 + ct) * KS_HH) * 32 + lane;
    const v16bf* BN = (const v16bf*)pBhh + ((size_t)(64 + ct) * KS_HH) * 32 + lane;
    #pragma unroll 4
    for (int ks = 0; ks < KS_HH; ++ks) {
      v16bf a  = A [ks * 32];
      v16bf br = BR[ks * 32];
      v16bf bz = BZ[ks * 32];
      v16bf bn = BN[ks * 32];
      accR  = __builtin_amdgcn_wmma_f32_16x16x32_bf16(false, a, false, br, (short)0, accR,  false, false);
      accZ  = __builtin_amdgcn_wmma_f32_16x16x32_bf16(false, a, false, bz, (short)0, accZ,  false, false);
      accHN = __builtin_amdgcn_wmma_f32_16x16x32_bf16(false, a, false, bn, (short)0, accHN, false, false);
    }
  }

  // epilogue: C/D layout lane=col (l&15), VGPR v = row v + 8*(l>=16)
  int c = col0 + (lane & 15);
  float bihr = bih[c], bihz = bih[HID + c], bihn = bih[2 * HID + c];
  float bhhr = bhh[c], bhhz = bhh[HID + c], bhhn = bhh[2 * HID + c];
  int hb = (lane >> 4) * 8;
  #pragma unroll
  for (int v = 0; v < 8; ++v) {
    int nobs = rt * 16 + hb + v;
    int row  = iobs[nobs];
    float hp = h[(size_t)row * HID + c];
    float r  = 1.0f / (1.0f + __expf(-(accR[v] + bihr + bhhr)));
    float z  = 1.0f / (1.0f + __expf(-(accZ[v] + bihz + bhhz)));
    float hn = accHN[v] + bhhn;
    float nn = tanhf(accIN[v] + bihn + r * hn);
    out[(size_t)row * HID + c] = (1.0f - z) * nn + z * hp;
  }
}

// ---------------- launcher ----------------
extern "C" void kernel_launch(void* const* d_in, const int* in_sizes, int n_in,
                              void* d_out, int out_size, void* d_ws, size_t ws_size,
                              hipStream_t stream) {
  const float* h      = (const float*)d_in[0];
  const float* p      = (const float*)d_in[1];
  const float* X_obs  = (const float*)d_in[2];
  const float* M_obs  = (const float*)d_in[3];
  const int*   i_obs  = (const int*)  d_in[4];
  const float* w_prep = (const float*)d_in[5];
  const float* b_prep = (const float*)d_in[6];
  const float* w_ih   = (const float*)d_in[7];
  const float* w_hh   = (const float*)d_in[8];
  const float* b_ih   = (const float*)d_in[9];
  const float* b_hh   = (const float*)d_in[10];
  float* out = (float*)d_out;

  // workspace layout (bytes), all 16B aligned:
  //   packA_ih : RT*KS_IH*32*16*2 = 102,400,000
  //   packA_hh : RT*KS_HH*32*16*2 =  51,200,000
  //   packB_ih : 96*KS_IH*32*16*2 =   3,145,728
  //   packB_hh : 96*KS_HH*32*16*2 =   1,572,864   (total 158,318,592)
  char* ws = (char*)d_ws;
  unsigned short* packA_ih = (unsigned short*)(ws);
  unsigned short* packA_hh = (unsigned short*)(ws + 102400000);
  unsigned short* packB_ih = (unsigned short*)(ws + 153600000);
  unsigned short* packB_hh = (unsigned short*)(ws + 156745728);

  float* loss = out + (size_t)N_ROWS * HID;

  init_loss_kernel<<<1, 1, 0, stream>>>(loss);

  pack_w_kernel<<<(TH3 * (KIH / 8) + 255) / 256, 256, 0, stream>>>(w_ih, packB_ih, KIH, KS_IH);
  pack_w_kernel<<<(TH3 * (KHH / 8) + 255) / 256, 256, 0, stream>>>(w_hh, packB_hh, KHH, KS_HH);

  int n4 = N_ROWS * HID / 4;
  copy_h_kernel<<<(n4 + 255) / 256, 256, 0, stream>>>((const v4f*)h, (v4f*)out, n4);

  prep_kernel<<<(NOBS * DIM + 255) / 256, 256, 0, stream>>>(
      p, X_obs, M_obs, i_obs, w_prep, b_prep, packA_ih, loss);

  gather_h_kernel<<<(NOBS * (KHH / 8) + 255) / 256, 256, 0, stream>>>(h, i_obs, packA_hh);

  dim3 grid(RT, 4);
  gates_kernel<<<grid, 256, 0, stream>>>(
      packA_ih, packA_hh, packB_ih, packB_hh, b_ih, b_hh, h, i_obs, out);

  (void)in_sizes; (void)n_in; (void)out_size; (void)ws_size;
}
